// EncodeContext_2826088481042
// MI455X (gfx1250) — compile-verified, matrix operand order checked by reading, not run on hardware
//
#include <hip/hip_runtime.h>

// ---------------- problem constants ----------------
#define BATCH   512
#define TSTEPS  100     // N_STEPS = TIME_STEP - 3
#define TFULL   103
#define FEAT    1024
#define HIDDEN  1024
#define KTOT    2048    // FEAT + HIDDEN (concatenated GEMM K)
#define NTOT    4096    // 4*HIDDEN

// ---------------- tiling ----------------
#define BM 128
#define BN 128          // = 4 gate strips of 32 columns
#define BK 32
#define GSTRIP 32       // per-gate column strip width per block
#define LDA 40          // padded LDS row stride (bf16 elems): 80B -> conflict-free frag reads
#define LDB 40
#define LDZ 132         // padded f32 stride for the epilogue tile

typedef __attribute__((ext_vector_type(16))) __bf16        v16bf;
typedef __attribute__((ext_vector_type(2)))  __bf16        v2bf;
typedef __attribute__((ext_vector_type(8)))  float         v8f;
typedef __attribute__((ext_vector_type(4)))  float         f32x4;
typedef __attribute__((ext_vector_type(4)))  unsigned int  u32x4;

union Frag { v16bf v; u32x4 q[2]; };

__device__ __forceinline__ unsigned pack2bf(float lo, float hi) {
    v2bf v;
    v[0] = (__bf16)lo;            // hardware v_cvt_pk_bf16_f32 (RNE)
    v[1] = (__bf16)hi;
    return __builtin_bit_cast(unsigned, v);
}
__device__ __forceinline__ unsigned short f2bfu(float f) {
    __bf16 b = (__bf16)f;
    return __builtin_bit_cast(unsigned short, b);
}

__device__ __forceinline__ float sigmoidf_(float x) {
    return 1.0f / (1.0f + __expf(-x));
}
__device__ __forceinline__ float tanhf_(float x) {
    float e = __expf(2.0f * x);
    return 1.0f - 2.0f / (e + 1.0f);
}

// ---- CDNA5 async global->LDS copy (ASYNCcnt-tracked DMA, bypasses VGPRs) ----
// Generic pointers to __shared__ carry the wave-relative LDS byte offset in
// their low 32 bits (ISA 10.2 aperture mapping), which is what the
// LDS-address VGPR of GLOBAL_LOAD_ASYNC_TO_LDS expects.
__device__ __forceinline__ void async_copy_b128(void* lds, const void* g) {
    unsigned ldsOff = (unsigned)(size_t)lds;
    asm volatile("global_load_async_to_lds_b128 %0, %1, off"
                 :: "v"(ldsOff), "v"((unsigned long long)(size_t)g)
                 : "memory");
}
__device__ __forceinline__ void wait_async0() {
    asm volatile("s_wait_asynccnt 0" ::: "memory");
}

// ---------------------------------------------------------------------------
// Prologue: [kernel ; recurrent_kernel] (f32, [2048,4096]) -> bf16 TRANSPOSED
// W2t[4096][2048] in ws.
// ---------------------------------------------------------------------------
__global__ void lstm_prep_kernel(const float* __restrict__ kw,
                                 const float* __restrict__ rw,
                                 unsigned short* __restrict__ W2t) {
    size_t i = (size_t)blockIdx.x * blockDim.x + threadIdx.x;  // 0 .. 8M-1
    const size_t half = (size_t)FEAT * NTOT;                   // 4M
    const int k = (int)(i >> 12);          // 0..2047
    const int n = (int)(i & (NTOT - 1));   // 0..4095
    float v = (i < half) ? kw[i] : rw[i - half];
    W2t[(size_t)n * KTOT + k] = f2bfu(v);
}

// ---------------------------------------------------------------------------
// Fused per-step kernel:
//   GEMM  Z = [x_t | h_{t-1}] (bf16) @ W2 (bf16), f32 accum
//   block N-range = 4 gate strips {j, j+1024, j+2048, j+3072}, j in [n0,n0+32)
//   epilogue: LSTM gate math -> C (ws), h (f32 -> d_out, bf16 -> Hbf)
// B tiles and the h-half of A are staged with async global->LDS DMA;
// only the x-half routes through VGPRs (needs f32->bf16 conversion).
// At t==0 the h-half is skipped entirely (h_{-1}==0) and c_old==0.
// ---------------------------------------------------------------------------
struct XRegs { f32x4 a[4]; };

__global__ __launch_bounds__(256)
void lstm_step_kernel(const float* __restrict__ xin,          // [512,103,1024]
                      const unsigned short* __restrict__ Hbf, // [512,1024] bf16 h_{t-1}
                      const unsigned short* __restrict__ W2t, // [4096,2048] bf16^T
                      const float* __restrict__ bias,         // [4096]
                      float* __restrict__ Cst,                // [512,1024] cell state
                      unsigned short* __restrict__ HbfOut,    // [512,1024] bf16 h_t
                      float* __restrict__ out,                // [512,100,1024]
                      int t) {
    __shared__ alignas(16) unsigned short As[2][BM * LDA];   // [m][k] bf16
    __shared__ alignas(16) unsigned short Bs[2][BN * LDB];   // [n][k] bf16 (K-contiguous)
    __shared__ alignas(16) float Zs[BM * LDZ];               // epilogue gate tile

    const int tid  = threadIdx.x;
    const int lane = tid & 31;
    const int wv   = tid >> 5;            // wave 0..7
    const int wm   = (wv & 1) * 64;       // wave row offset in tile
    const int wn   = (wv >> 1) * 32;      // wave col offset in tile
    const int lr   = lane & 15;
    const int hi   = lane >> 4;           // 0 / 1
    const int ksel = hi ? 8 : 0;          // K-half select per ISA 16-bit layout

    const int n0  = blockIdx.x * GSTRIP;  // gate-strip base j (0..992)
    const int bm0 = blockIdx.y * BM;      // global row base

    // ---- staging thread mapping ----
    const int am   = tid >> 2;            // A row 0..63  (and +64)
    const int akq  = (tid & 3) * 8;       // A k offset: 0,8,16,24 (8 elems)
    const int bnr  = tid >> 2;            // B tile row 0..63 (and +64)
    const int bseg = (tid & 3) * 8;       // B k offset: 8 bf16 = 16B

    // tile n -> global weight column: (n/32)*1024 + n0 + (n%32)
    const int colLo = ((bnr >> 5) + 0) * HIDDEN + n0 + (bnr & 31);  // strips 0,1
    const int colHi = ((bnr >> 5) + 2) * HIDDEN + n0 + (bnr & 31);  // strips 2,3

    const float* aX  = xin + (size_t)(bm0 + am) * (TFULL * FEAT) + (size_t)t * FEAT + akq;
    const float* aX2 = aX + (size_t)64 * (TFULL * FEAT);
    const unsigned short* aH  = Hbf + (size_t)(bm0 + am) * HIDDEN + akq;
    const unsigned short* aH2 = aH + (size_t)64 * HIDDEN;
    const unsigned short* bW  = W2t + (size_t)colLo * KTOT + bseg;
    const unsigned short* bW2 = W2t + (size_t)colHi * KTOT + bseg;

    const int kEnd = (t == 0) ? FEAT : KTOT;   // h_{-1}==0 -> skip recurrent half

    v8f acc[4][2] = {};

    // ---- pipeline helpers ----
    auto issue_async_stage = [&](int buf, int kc) {
        // B tile: pure bf16 copy -> async DMA
        async_copy_b128(&Bs[buf][bnr * LDB + bseg],        bW  + kc);
        async_copy_b128(&Bs[buf][(bnr + 64) * LDB + bseg], bW2 + kc);
        if (kc >= FEAT) {   // h-half of A: pure bf16 copy -> async DMA
            async_copy_b128(&As[buf][am * LDA + akq],        aH  + (kc - FEAT));
            async_copy_b128(&As[buf][(am + 64) * LDA + akq], aH2 + (kc - FEAT));
        }
    };
    auto load_x = [&](int kc) -> XRegs {      // x-half of A: f32, convert later
        XRegs r;
        r.a[0] = *(const f32x4*)(aX + kc);
        r.a[1] = *(const f32x4*)(aX + kc + 4);
        r.a[2] = *(const f32x4*)(aX2 + kc);
        r.a[3] = *(const f32x4*)(aX2 + kc + 4);
        return r;
    };
    auto store_x = [&](int buf, const XRegs& r) {
        u32x4 p0, p1;
        p0.x = pack2bf(r.a[0].x, r.a[0].y); p0.y = pack2bf(r.a[0].z, r.a[0].w);
        p0.z = pack2bf(r.a[1].x, r.a[1].y); p0.w = pack2bf(r.a[1].z, r.a[1].w);
        p1.x = pack2bf(r.a[2].x, r.a[2].y); p1.y = pack2bf(r.a[2].z, r.a[2].w);
        p1.z = pack2bf(r.a[3].x, r.a[3].y); p1.w = pack2bf(r.a[3].z, r.a[3].w);
        *(u32x4*)&As[buf][am * LDA + akq]        = p0;
        *(u32x4*)&As[buf][(am + 64) * LDA + akq] = p1;
    };
    auto compute = [&](int buf) {
        Frag bfrg[2];
        #pragma unroll
        for (int ni = 0; ni < 2; ++ni) {
            const unsigned short* bp = &Bs[buf][(wn + ni * 16 + lr) * LDB + ksel];
            bfrg[ni].q[0] = *(const u32x4*)bp;
            bfrg[ni].q[1] = *(const u32x4*)(bp + 16);
        }
        #pragma unroll
        for (int mi = 0; mi < 4; ++mi) {
            Frag af;
            const unsigned short* ap = &As[buf][(wm + mi * 16 + lr) * LDA + ksel];
            af.q[0] = *(const u32x4*)ap;
            af.q[1] = *(const u32x4*)(ap + 16);
            #pragma unroll
            for (int ni = 0; ni < 2; ++ni)
                acc[mi][ni] = __builtin_amdgcn_wmma_f32_16x16x32_bf16(
                    false, af.v, false, bfrg[ni].v,
                    (short)0, acc[mi][ni], false, false);
        }
    };

    // ---- software pipeline: stage(k+1) overlaps wmma(k) ----
    {
        issue_async_stage(0, 0);
        XRegs r = load_x(0);              // kc==0 is always the x half
        store_x(0, r);
        wait_async0();
    }
    __syncthreads();

    int buf = 0;
    for (int kc = BK; kc < kEnd; kc += BK) {
        issue_async_stage(buf ^ 1, kc);             // async DMA for next chunk
        const bool isX = kc < FEAT;
        XRegs r;
        if (isX) r = load_x(kc);                    // issue x global loads early
        __builtin_prefetch(bW + kc + BK, 0, 3);     // prefetch next weight chunk near
        compute(buf);                               // WMMA on current buffer
        if (isX) store_x(buf ^ 1, r);               // convert + LDS store
        wait_async0();                              // this wave's DMA done
        __syncthreads();
        buf ^= 1;
    }
    compute(buf);

    // ---- epilogue: accumulators -> LDS, then fused LSTM gate math ----
    #pragma unroll
    for (int mi = 0; mi < 4; ++mi)
        #pragma unroll
        for (int ni = 0; ni < 2; ++ni)
            #pragma unroll
            for (int r = 0; r < 8; ++r) {
                const int row  = wm + mi * 16 + hi * 8 + r;
                const int colt = wn + ni * 16 + lr;
                Zs[row * LDZ + colt] = acc[mi][ni][r];
            }
    __syncthreads();

    {
        const int j  = tid & 31;                 // strip-local column
        const int rg = (tid >> 5) * 16;          // row group base (0,16,...,112)
        const int jg = n0 + j;                   // global hidden index
        const float bi = bias[jg];
        const float bf = bias[HIDDEN + jg];
        const float bg = bias[2 * HIDDEN + jg];
        const float bo = bias[3 * HIDDEN + jg];
        #pragma unroll
        for (int rr = 0; rr < 16; ++rr) {
            const int row = rg + rr;
            const int gb  = bm0 + row;           // batch index
            const float zi = Zs[row * LDZ + j]      + bi;
            const float zf = Zs[row * LDZ + 32 + j] + bf;
            const float zg = Zs[row * LDZ + 64 + j] + bg;
            const float zo = Zs[row * LDZ + 96 + j] + bo;
            const size_t ci = (size_t)gb * HIDDEN + jg;
            const float c_old = (t == 0) ? 0.0f : Cst[ci];
            const float c_new = sigmoidf_(zf) * c_old + sigmoidf_(zi) * tanhf_(zg);
            const float h_new = sigmoidf_(zo) * tanhf_(c_new);
            Cst[ci] = c_new;
            HbfOut[ci] = f2bfu(h_new);
            out[(size_t)gb * (TSTEPS * HIDDEN) + (size_t)t * HIDDEN + jg] = h_new;
        }
    }
}

// ---------------------------------------------------------------------------
extern "C" void kernel_launch(void* const* d_in, const int* in_sizes, int n_in,
                              void* d_out, int out_size, void* d_ws, size_t ws_size,
                              hipStream_t stream) {
    const float* xin  = (const float*)d_in[0];   // [512,103,1024]
    const float* kw   = (const float*)d_in[1];   // [1024,4096]
    const float* rw   = (const float*)d_in[2];   // [1024,4096]
    const float* bias = (const float*)d_in[3];   // [4096]
    float* out = (float*)d_out;                  // [512,100,1024]

    // workspace: W2t bf16 (16MB) | C f32 (2MB) | Hbf bf16 (1MB)
    unsigned short* W2t = (unsigned short*)d_ws;
    float* Cst = (float*)((char*)d_ws + (size_t)KTOT * NTOT * sizeof(unsigned short));
    unsigned short* Hbf = (unsigned short*)(Cst + (size_t)BATCH * HIDDEN);

    // prologue: weight convert + transpose (8M threads)
    {
        const size_t total = (size_t)KTOT * NTOT;   // 8M
        lstm_prep_kernel<<<dim3((unsigned)(total / 256)), dim3(256), 0, stream>>>(
            kw, rw, W2t);
    }

    dim3 ggrid(HIDDEN / GSTRIP, BATCH / BM);     // 32 x 4 = 128 blocks
    dim3 gblk(256);
    for (int t = 0; t < TSTEPS; ++t) {
        lstm_step_kernel<<<ggrid, gblk, 0, stream>>>(
            xin, Hbf, W2t, bias, Cst, Hbf, out, t);
    }
}
